// SpeechLMTransformer_26456998543937
// MI455X (gfx1250) — compile-verified, matrix operand order checked by reading, not run on hardware
//
#include <hip/hip_runtime.h>
#include <hip/hip_bf16.h>
#include <cstdint>

// ---------------------------------------------------------------------------
// CDNA5 (gfx1250) wave32 WMMA transformer forward pass.
// GEMM A-tiles are DMA'd by the Tensor Data Mover (tensor_load_to_lds) with
// double buffering on TENSORcnt; B-tiles stream through float4 loads with
// global_prefetch; all matmuls use v_wmma_f32_16x16x32_f16.
// ---------------------------------------------------------------------------

typedef __attribute__((ext_vector_type(16))) _Float16 v16h;
typedef __attribute__((ext_vector_type(8)))  float    v8f;
typedef __attribute__((ext_vector_type(4)))  unsigned int v4u;
typedef __attribute__((ext_vector_type(8)))  int      v8i;
typedef __attribute__((ext_vector_type(4)))  int      v4i;

__device__ __forceinline__ v8f wmma_f16(v16h a, v16h b, v8f c) {
  // D(16x16,f32) = A(16x32,f16) * B(32x16,f16) + C
  return __builtin_amdgcn_wmma_f32_16x16x32_f16(false, a, false, b,
                                                (short)0, c, false, false);
}

__device__ __forceinline__ float gelu_f(float x) {
  return 0.5f * x * (1.0f + erff(x * 0.70710678118654752f));
}

// ---------------------------------------------------------------------------
// TDM: DMA a 2D f32 tile [rows x cols] (row stride = strideElems) from global
// into LDS at ldsAddr, with LDS padding of 4 DWORDs after every 32 DWORDs
// (so a 32-float tile row occupies 36 floats in LDS -> conflict-free reads).
// Descriptor layout per CDNA5 ISA section 8.3/8.4 (D# groups 0 and 1).
// This toolchain (clang-23) exposes the 6-arg builtin form; the extra group
// slot is zero-filled (the D# spec defines only 4 groups for <=5D tensors).
// ---------------------------------------------------------------------------
__device__ __forceinline__ void tdm_load_tile_f32(const float* gsrc,
                                                  unsigned ldsAddr,
                                                  int strideElems,
                                                  int rows, int cols)
{
  unsigned long long ga = (unsigned long long)(uintptr_t)gsrc;
  v4u g0;
  g0[0] = 1u;                                            // count=1, user D#
  g0[1] = ldsAddr;                                       // lds_addr (bytes)
  g0[2] = (unsigned)(ga & 0xFFFFFFFFu);                  // global_addr[31:0]
  g0[3] = (unsigned)((ga >> 32) & 0x01FFFFFFu)           // global_addr[56:32]
        | 0x80000000u;                                   // type=2 ("image")
  unsigned td0 = (unsigned)strideElems;                  // tensor_dim0
  unsigned td1 = 0x10000u;                               // tensor_dim1 (ample)
  v8i g1;
  g1[0] = (int)((2u << 16)                               // data_size = 4B
              | (1u << 20)                               // pad_enable
              | (4u << 22)                               // pad_interval: 32 DW
              | (3u << 25));                             // pad_amount: 4 DW
  g1[1] = (int)((td0 & 0xFFFFu) << 16);                  // dim0[15:0]
  g1[2] = (int)(((td0 >> 16) & 0xFFFFu)                  // dim0[31:16]
              | ((td1 & 0xFFFFu) << 16));                // dim1[15:0]
  g1[3] = (int)(((td1 >> 16) & 0xFFFFu)                  // dim1[31:16]
              | ((unsigned)cols << 16));                 // tile_dim0
  g1[4] = rows;                                          // tile_dim1 (dim2=0)
  g1[5] = (int)td0;                                      // dim0_stride[31:0]
  g1[6] = 0;                                             // stride hi bits
  g1[7] = 0;
  v4i gz = {0, 0, 0, 0};                                 // groups 2/3 unused
  v8i gz8 = {0, 0, 0, 0, 0, 0, 0, 0};                    // extra slot (zero)
  __builtin_amdgcn_tensor_load_to_lds(g0, g1, gz, gz, gz8, 0);
}

// ---------------------------------------------------------------------------
// Tiled WMMA GEMM: C[M,N] = A[M,K] * W[K,N]  (+bias) (+gelu | +residual)
// Requires: N % 64 == 0, K % 32 == 0 (true for all uses). M guarded only on
// edge tiles (TDM fast path for interior tiles).
// Block = 128 threads (4 waves); each wave owns a 32x32 quadrant of a 64x64
// tile (2x2 of 16x16 WMMA tiles).
// epi: 0 = bias only, 1 = bias+GELU, 2 = bias+residual
// ---------------------------------------------------------------------------
#define TBM 64
#define TBN 64
#define TBK 32
#define AROW 36            // 32 floats + 4 floats TDM pad
#define LPAD 8

__global__ __launch_bounds__(128)
void gemm_wmma_kernel(const float* __restrict__ A, const float* __restrict__ W,
                      const float* __restrict__ bias, const float* __restrict__ resid,
                      float* __restrict__ C, int M, int N, int K, int epi)
{
  __shared__ float    Af32[2][TBM][AROW];    // TDM destination, double-buffered
  __shared__ _Float16 Bs[TBN][TBK + LPAD];   // [n][k] (transposed on store)

  const int tid  = threadIdx.x;
  const int lane = tid & 31;
  const int wv   = tid >> 5;
  const int lr   = lane & 15;
  const int hi   = lane >> 4;
  const int wr   = (wv >> 1) << 5;     // wave row offset: 0 / 32
  const int wc   = (wv & 1) << 5;      // wave col offset: 0 / 32
  const int mBase = blockIdx.y * TBM;
  const int nBase = blockIdx.x * TBN;
  const bool interior = (mBase + TBM) <= M;
  const int nK = K / TBK;

  const v8f vzero = {0.f,0.f,0.f,0.f,0.f,0.f,0.f,0.f};
  v8f acc[2][2];
  acc[0][0] = vzero; acc[0][1] = vzero; acc[1][0] = vzero; acc[1][1] = vzero;

  if (interior && wv == 0) {
    tdm_load_tile_f32(A + (size_t)mBase * K,
                      (unsigned)(uintptr_t)&Af32[0][0][0], K, TBM, TBK);
  }

  for (int kt = 0; kt < nK; ++kt) {
    const int k0 = kt * TBK;
    const int buf = interior ? (kt & 1) : 0;

    if (interior) {
      if (wv == 0 && (kt + 1) < nK) {
        tdm_load_tile_f32(A + (size_t)mBase * K + (k0 + TBK),
                          (unsigned)(uintptr_t)&Af32[(kt + 1) & 1][0][0],
                          K, TBM, TBK);
      }
    } else {
      // Edge tiles (rare): guarded scalar loads into buffer 0.
      #pragma unroll
      for (int i = 0; i < 16; ++i) {
        int idx = tid + (i << 7);
        int r = idx >> 5, c = idx & 31;
        int gr = mBase + r;
        Af32[0][r][c] = (gr < M) ? A[(size_t)gr * K + (k0 + c)] : 0.f;
      }
    }

    // B tile: unguarded float4 loads, stored transposed (K-pairs contiguous)
    #pragma unroll
    for (int i = 0; i < 4; ++i) {
      int r  = (tid >> 4) + (i << 3);            // k row 0..31
      int c4 = (tid & 15) << 2;                  // 0,4,...,60
      const float4 vb4 = *(const float4*)&W[(size_t)(k0 + r) * N + nBase + c4];
      Bs[c4 + 0][r] = (_Float16)vb4.x;
      Bs[c4 + 1][r] = (_Float16)vb4.y;
      Bs[c4 + 2][r] = (_Float16)vb4.z;
      Bs[c4 + 3][r] = (_Float16)vb4.w;
      if ((k0 + TBK) < K)                         // prefetch next B tile
        __builtin_prefetch(&W[(size_t)(k0 + TBK + r) * N + nBase + c4], 0, 0);
    }

    if (interior && wv == 0) {
      if ((kt + 1) < nK) __builtin_amdgcn_s_wait_tensorcnt(1);
      else               __builtin_amdgcn_s_wait_tensorcnt(0);
    }
    __syncthreads();

    // Fragments per documented 16-bit A/B layout:
    // vgpr j holds K = (j>>2)*16 + hi*8 + (j&3)*2 (+1)
    v16h af[2], bf[2];
    #pragma unroll
    for (int t = 0; t < 2; ++t) {
      const float*    pa = &Af32[buf][wr + (t << 4) + lr][0];
      const _Float16* pb = &Bs[wc + (t << 4) + lr][0];
      v16h a, b;
      #pragma unroll
      for (int j = 0; j < 8; ++j) {
        int ks = ((j >> 2) << 4) + (hi << 3) + ((j & 3) << 1);
        a[2*j]   = (_Float16)pa[ks];  a[2*j+1] = (_Float16)pa[ks + 1];
        b[2*j]   = pb[ks];            b[2*j+1] = pb[ks + 1];
      }
      af[t] = a; bf[t] = b;
    }
    acc[0][0] = wmma_f16(af[0], bf[0], acc[0][0]);
    acc[0][1] = wmma_f16(af[0], bf[1], acc[0][1]);
    acc[1][0] = wmma_f16(af[1], bf[0], acc[1][0]);
    acc[1][1] = wmma_f16(af[1], bf[1], acc[1][1]);
    __syncthreads();
  }

  // Epilogue. C layout: lanes 0-15 -> M = r, lanes 16-31 -> M = r+8; N = lr.
  #pragma unroll
  for (int tm = 0; tm < 2; ++tm)
    #pragma unroll
    for (int tn = 0; tn < 2; ++tn)
      #pragma unroll
      for (int r = 0; r < 8; ++r) {
        int grow = mBase + wr + (tm << 4) + r + (hi << 3);
        int gcol = nBase + wc + (tn << 4) + lr;
        if (interior || grow < M) {
          float v = acc[tm][tn][r];
          if (bias)     v += bias[gcol];
          if (epi == 1) v = gelu_f(v);
          if (epi == 2) v += resid[(size_t)grow * N + gcol];
          C[(size_t)grow * N + gcol] = v;
        }
      }
}

// ---------------------------------------------------------------------------
// LayerNorm over D=1024, one 256-thread block per row.
// ---------------------------------------------------------------------------
__global__ __launch_bounds__(256)
void layernorm_kernel(const float* __restrict__ x, const float* __restrict__ g,
                      const float* __restrict__ b, float* __restrict__ y,
                      long inStride, long outStride)
{
  const int row = blockIdx.x;
  const float* xr = x + (size_t)row * inStride;
  float* yr = y + (size_t)row * outStride;
  const int tid = threadIdx.x;

  float v[4], s = 0.f, s2 = 0.f;
  #pragma unroll
  for (int i = 0; i < 4; ++i) {
    v[i] = xr[tid + (i << 8)];
    s += v[i]; s2 += v[i] * v[i];
  }
  #pragma unroll
  for (int off = 16; off >= 1; off >>= 1) {
    s  += __shfl_xor(s,  off, 32);
    s2 += __shfl_xor(s2, off, 32);
  }
  __shared__ float sm[8], sv[8], stat[2];
  if ((tid & 31) == 0) { sm[tid >> 5] = s; sv[tid >> 5] = s2; }
  __syncthreads();
  if (tid == 0) {
    float ts = 0.f, ts2 = 0.f;
    #pragma unroll
    for (int i = 0; i < 8; ++i) { ts += sm[i]; ts2 += sv[i]; }
    float mean = ts * (1.f / 1024.f);
    float var  = ts2 * (1.f / 1024.f) - mean * mean;   // population var
    stat[0] = mean; stat[1] = rsqrtf(var + 1e-5f);
  }
  __syncthreads();
  float mean = stat[0], rstd = stat[1];
  #pragma unroll
  for (int i = 0; i < 4; ++i) {
    int c = tid + (i << 8);
    yr[c] = (v[i] - mean) * rstd * g[c] + b[c];
  }
}

// ---------------------------------------------------------------------------
// Embedding gather (D=1024): dst row = (row/rpb)*batchStride + (row%rpb)*1024
//                                      + rowOffset
// ---------------------------------------------------------------------------
__global__ __launch_bounds__(256)
void gather_kernel(const int* __restrict__ tok, const float* __restrict__ table,
                   float* __restrict__ out, int rowsPerBatch, long batchStride,
                   long rowOffset)
{
  const int row = blockIdx.x;
  const int t = tok[row];
  const float* src = table + (size_t)t * 1024;
  float* dst = out + (size_t)(row / rowsPerBatch) * batchStride
                   + (size_t)(row % rowsPerBatch) * 1024 + rowOffset;
  for (int i = threadIdx.x; i < 1024; i += 256) dst[i] = src[i];
}

// ---------------------------------------------------------------------------
// RoPE applied in place to q and k slices of qkv[B,S,3*1024].
// For head h, pair i (0..31): [x1,x2]=[d=i, d=i+32]; theta = s * 10000^(-i/32).
// ---------------------------------------------------------------------------
__global__ __launch_bounds__(256)
void rope_kernel(float* __restrict__ qkv)
{
  const int token = blockIdx.x;          // b*1024 + s
  const int s = token & 1023;
  float* base = qkv + (size_t)token * 3072;
  for (int p = threadIdx.x; p < 1024; p += 256) {
    const int which = p >> 9;            // 0=q, 1=k
    const int pp = p & 511;
    const int h  = pp >> 5;
    const int i  = pp & 31;
    float inv = __expf(-(float)i * 0.28782313662425572f);  // ln(10000)/32
    float fr = (float)s * inv;
    float sn, cs;
    __sincosf(fr, &sn, &cs);
    float* ptr = base + (which << 10) + (h << 6);
    float x1 = ptr[i], x2 = ptr[i + 32];
    ptr[i]      = x1 * cs - x2 * sn;
    ptr[i + 32] = x2 * cs + x1 * sn;
  }
}

// ---------------------------------------------------------------------------
// WMMA flash attention. One wave (32 threads) per (query tile of 16, b, h).
// HD = 64. scores = Q*K^T via 2 WMMAs; online softmax; P*V via 4 WMMAs with
// K zero-padded to 32. Out-of-range KV columns handled by clamped addresses
// plus -1e30 score masking (their P entries become exactly 0), so all global
// loads are unconditional (mergeable into wide loads).
// ---------------------------------------------------------------------------
__global__ __launch_bounds__(32)
void attn_kernel(const float* __restrict__ q, long qRowStride, long qBatchStride,
                 const float* __restrict__ k, const float* __restrict__ v,
                 long kvRowStride, long kvBatchStride, int kvLen,
                 float* __restrict__ o, long oRowStride, long oBatchStride,
                 int causal)
{
  __shared__ _Float16 Ps[16][24];        // P transpose staging (padded)
  const int lane = threadIdx.x;
  const int lr = lane & 15, hi = lane >> 4;
  const int qt = blockIdx.x;
  const int bh = blockIdx.y;
  const int b = bh >> 4, h = bh & 15;
  const int hd0 = h << 6;
  const int m0 = qt << 4;
  const float scale = 0.125f;            // 1/sqrt(64)

  const float* qb = q + (size_t)b * qBatchStride + hd0;
  const float* kb = k + (size_t)b * kvBatchStride + hd0;
  const float* vb = v + (size_t)b * kvBatchStride + hd0;

  // Q fragments: A-layout, row = m0+lr, two 32-wide K halves over HD
  const float* qrow = qb + (size_t)(m0 + lr) * qRowStride;
  v16h qa[2];
  #pragma unroll
  for (int half = 0; half < 2; ++half) {
    v16h a;
    #pragma unroll
    for (int j = 0; j < 8; ++j) {
      int ks = (half << 5) + ((j >> 2) << 4) + (hi << 3) + ((j & 3) << 1);
      a[2*j]   = (_Float16)qrow[ks];
      a[2*j+1] = (_Float16)qrow[ks + 1];
    }
    qa[half] = a;
  }

  const v8f vzero = {0.f,0.f,0.f,0.f,0.f,0.f,0.f,0.f};
  float mrow[8], lrow[8];
  v8f acc[4];
  #pragma unroll
  for (int r = 0; r < 8; ++r) { mrow[r] = -1e30f; lrow[r] = 0.f; }
  #pragma unroll
  for (int t = 0; t < 4; ++t) acc[t] = vzero;

  const int nTiles = causal ? (qt + 1) : ((kvLen + 15) >> 4);
  for (int kt = 0; kt < nTiles; ++kt) {
    const int kc0 = kt << 4;
    const int kcol = kc0 + lr;
    const int kcolc = min(kcol, kvLen - 1);            // clamped for loads
    const float* krow = kb + (size_t)kcolc * kvRowStride;

    // K^T fragments: B-layout, col = key, element = head dim (unconditional)
    v16h kf[2];
    #pragma unroll
    for (int half = 0; half < 2; ++half) {
      v16h bfr;
      #pragma unroll
      for (int j = 0; j < 8; ++j) {
        int ks = (half << 5) + ((j >> 2) << 4) + (hi << 3) + ((j & 3) << 1);
        bfr[2*j]   = (_Float16)krow[ks];
        bfr[2*j+1] = (_Float16)krow[ks + 1];
      }
      kf[half] = bfr;
    }
    v8f s8 = wmma_f16(qa[0], kf[0], vzero);
    s8 = wmma_f16(qa[1], kf[1], s8);

    // scale + masks BEFORE reductions (kills clamped-load garbage)
    #pragma unroll
    for (int r = 0; r < 8; ++r) {
      float sv = s8[r] * scale;
      int qg = m0 + r + (hi << 3);
      if ((causal && (kcol > qg)) || (kcol >= kvLen)) sv = -1e30f;
      s8[r] = sv;
    }

    // online softmax: reductions across the 16 lanes that own each row
    #pragma unroll
    for (int r = 0; r < 8; ++r) {
      float mx = s8[r];
      #pragma unroll
      for (int off = 8; off >= 1; off >>= 1)
        mx = fmaxf(mx, __shfl_xor(mx, off, 32));
      float mnew = fmaxf(mrow[r], mx);
      float p = __expf(s8[r] - mnew);
      float ps = p;
      #pragma unroll
      for (int off = 8; off >= 1; off >>= 1)
        ps += __shfl_xor(ps, off, 32);
      float corr = __expf(mrow[r] - mnew);
      lrow[r] = lrow[r] * corr + ps;
      mrow[r] = mnew;
      s8[r] = p;
      #pragma unroll
      for (int t = 0; t < 4; ++t) acc[t][r] *= corr;
    }

    // transpose P (C layout) -> A layout via LDS (wave-local, LDS in-order)
    #pragma unroll
    for (int r = 0; r < 8; ++r) Ps[r + (hi << 3)][lr] = (_Float16)s8[r];

    v16h pf;
    #pragma unroll
    for (int j = 0; j < 8; ++j) {
      if (j < 4) {
        int ks = (hi << 3) + (j << 1);       // K 0..15 real
        pf[2*j]   = Ps[lr][ks];
        pf[2*j+1] = Ps[lr][ks + 1];
      } else {                               // K 16..31 zero pad
        pf[2*j] = (_Float16)0.f; pf[2*j+1] = (_Float16)0.f;
      }
    }

    // P*V: 4 output tiles of 16 dims each; clamped rows (P==0 kills them)
    #pragma unroll
    for (int t = 0; t < 4; ++t) {
      v16h vf;
      int dim = (t << 4) + lr;
      #pragma unroll
      for (int j = 0; j < 8; ++j) {
        if (j < 4) {
          int ks = (hi << 3) + (j << 1);
          int key0 = min(kc0 + ks,     kvLen - 1);
          int key1 = min(kc0 + ks + 1, kvLen - 1);
          vf[2*j]   = (_Float16)vb[(size_t)key0 * kvRowStride + dim];
          vf[2*j+1] = (_Float16)vb[(size_t)key1 * kvRowStride + dim];
        } else {
          vf[2*j] = (_Float16)0.f; vf[2*j+1] = (_Float16)0.f;
        }
      }
      acc[t] = wmma_f16(pf, vf, acc[t]);
    }
  }

  float* ob = o + (size_t)b * oBatchStride + hd0;
  #pragma unroll
  for (int t = 0; t < 4; ++t)
    #pragma unroll
    for (int r = 0; r < 8; ++r) {
      int qg = m0 + r + (hi << 3);
      ob[(size_t)qg * oRowStride + (t << 4) + lr] = acc[t][r] / lrow[r];
    }
}

// ---------------------------------------------------------------------------
// Host-side orchestration (graph-capture safe: only kernel launches).
// ---------------------------------------------------------------------------
extern "C" void kernel_launch(void* const* d_in, const int* in_sizes, int n_in,
                              void* d_out, int out_size, void* d_ws, size_t ws_size,
                              hipStream_t stream)
{
  (void)in_sizes; (void)n_in; (void)out_size; (void)ws_size;

  const int*   audio_tokens  = (const int*)d_in[0];
  const int*   text_tokens   = (const int*)d_in[1];
  const float* speaker_emb   = (const float*)d_in[2];
  const float* token_embed_w = (const float*)d_in[3];
  const float* text_embed_w  = (const float*)d_in[4];
  const float* spk_w    = (const float*)d_in[5];
  const float* spk_b    = (const float*)d_in[6];
  const float* spk_ln_g = (const float*)d_in[7];
  const float* spk_ln_b = (const float*)d_in[8];
  const float* norm1_g  = (const float*)d_in[9];
  const float* norm1_b  = (const float*)d_in[10];
  const float* qkv_w    = (const float*)d_in[11];
  const float* out_w    = (const float*)d_in[12];
  const float* out_b    = (const float*)d_in[13];
  const float* normc_g  = (const float*)d_in[14];
  const float* normc_b  = (const float*)d_in[15];
  const float* cq_w     = (const float*)d_in[16];
  const float* ckv_w    = (const float*)d_in[17];
  const float* cout_w   = (const float*)d_in[18];
  const float* cout_b   = (const float*)d_in[19];
  const float* norm2_g  = (const float*)d_in[20];
  const float* norm2_b  = (const float*)d_in[21];
  const float* ffn1_w   = (const float*)d_in[22];
  const float* ffn1_b   = (const float*)d_in[23];
  const float* ffn2_w   = (const float*)d_in[24];
  const float* ffn2_b   = (const float*)d_in[25];
  const float* normout_g = (const float*)d_in[26];
  const float* normout_b = (const float*)d_in[27];
  const float* lm_head_w = (const float*)d_in[28];

  // Workspace carve-up (floats). Total ~21.77M floats (~87 MB).
  float* ws   = (float*)d_ws;
  float* x    = ws;  ws += 2097152;                 // [2,1024,1024]
  float* hbuf = ws;  ws += 2097152;                 // [2,1024,1024]
  float* qkv  = ws;  ws += 6291456;                 // [2,1024,3072]
  float* obuf = ws;  ws += 2097152;                 // [2,1024,1024]
  float* kvc  = ws;  ws += 528384;                  // [2,129,2048]
  float* cond = ws;  ws += 264192;                  // [2,129,1024]
  float* ffnh = ws;  ws += 8388608;                 // [2,1024,4096]
  float* spk  = ws;  ws += 2048;                    // [2,1024]
  float* qc   = qkv;                                // reuse qkv region

  const dim3 blk128(128), blk256(256), blk32(32);

  // x = token_embed_w[audio_tokens]
  gather_kernel<<<dim3(2048), blk256, 0, stream>>>(
      audio_tokens, token_embed_w, x, 1024, 1048576L, 0L);
  // spk = gelu(speaker_emb @ spk_w + spk_b)
  gemm_wmma_kernel<<<dim3(16, 1), blk128, 0, stream>>>(
      speaker_emb, spk_w, spk_b, nullptr, spk, 2, 1024, 256, 1);
  // cond[:,0,:] = LN(spk)
  layernorm_kernel<<<dim3(2), blk256, 0, stream>>>(
      spk, spk_ln_g, spk_ln_b, cond, 1024L, 132096L);
  // cond[:,1:,:] = text_embed_w[text_tokens]
  gather_kernel<<<dim3(256), blk256, 0, stream>>>(
      text_tokens, text_embed_w, cond, 128, 132096L, 1024L);

  for (int i = 0; i < 6; ++i) {
    // ---- self attention ----
    layernorm_kernel<<<dim3(2048), blk256, 0, stream>>>(
        x, norm1_g + (size_t)i*1024, norm1_b + (size_t)i*1024, hbuf, 1024L, 1024L);
    gemm_wmma_kernel<<<dim3(48, 32), blk128, 0, stream>>>(
        hbuf, qkv_w + (size_t)i*3145728, nullptr, nullptr, qkv, 2048, 3072, 1024, 0);
    rope_kernel<<<dim3(2048), blk256, 0, stream>>>(qkv);
    attn_kernel<<<dim3(64, 32), blk32, 0, stream>>>(
        qkv, 3072L, 3145728L, qkv + 1024, qkv + 2048, 3072L, 3145728L, 1024,
        obuf, 1024L, 1048576L, 1);
    gemm_wmma_kernel<<<dim3(16, 32), blk128, 0, stream>>>(
        obuf, out_w + (size_t)i*1048576, out_b + (size_t)i*1024, x, x,
        2048, 1024, 1024, 2);
    // ---- cross attention ----
    layernorm_kernel<<<dim3(2048), blk256, 0, stream>>>(
        x, normc_g + (size_t)i*1024, normc_b + (size_t)i*1024, hbuf, 1024L, 1024L);
    gemm_wmma_kernel<<<dim3(16, 32), blk128, 0, stream>>>(
        hbuf, cq_w + (size_t)i*1048576, nullptr, nullptr, qc, 2048, 1024, 1024, 0);
    gemm_wmma_kernel<<<dim3(32, 5), blk128, 0, stream>>>(
        cond, ckv_w + (size_t)i*2097152, nullptr, nullptr, kvc, 258, 2048, 1024, 0);
    attn_kernel<<<dim3(64, 32), blk32, 0, stream>>>(
        qc, 1024L, 1048576L, kvc, kvc + 1024, 2048L, 264192L, 129,
        obuf, 1024L, 1048576L, 0);
    gemm_wmma_kernel<<<dim3(16, 32), blk128, 0, stream>>>(
        obuf, cout_w + (size_t)i*1048576, cout_b + (size_t)i*1024, x, x,
        2048, 1024, 1024, 2);
    // ---- FFN ----
    layernorm_kernel<<<dim3(2048), blk256, 0, stream>>>(
        x, norm2_g + (size_t)i*1024, norm2_b + (size_t)i*1024, hbuf, 1024L, 1024L);
    gemm_wmma_kernel<<<dim3(64, 32), blk128, 0, stream>>>(
        hbuf, ffn1_w + (size_t)i*4194304, ffn1_b + (size_t)i*4096, nullptr, ffnh,
        2048, 4096, 1024, 1);
    gemm_wmma_kernel<<<dim3(16, 32), blk128, 0, stream>>>(
        ffnh, ffn2_w + (size_t)i*4194304, ffn2_b + (size_t)i*1024, x, x,
        2048, 1024, 4096, 2);
  }

  // final LN + lm_head
  layernorm_kernel<<<dim3(2048), blk256, 0, stream>>>(
      x, normout_g, normout_b, hbuf, 1024L, 1024L);
  gemm_wmma_kernel<<<dim3(16, 32), blk128, 0, stream>>>(
      hbuf, lm_head_w, nullptr, nullptr, (float*)d_out, 2048, 1024, 1024, 0);
}
